// GatedDuoAttention_76751065580074
// MI455X (gfx1250) — matrix operand order repaired and checked by reading, not compiled, not run on hardware
//
#include <hip/hip_runtime.h>

// ---------------- problem constants ----------------
constexpr int S   = 2048;
constexpr int H   = 2048;
constexpr int NH  = 16;
constexpr int R   = 4;
constexpr int KVH = NH / R;      // 4
constexpr int HD  = H / NH;      // 128
constexpr int KVW = KVH * HD;    // 512
constexpr int WINH = 64;         // WIN/2

// ---------------- types ----------------
typedef __attribute__((ext_vector_type(16))) __bf16        v16bf;
typedef __attribute__((ext_vector_type(8)))  float         v8f;
typedef __attribute__((ext_vector_type(16))) unsigned short v16u;
typedef __attribute__((ext_vector_type(4)))  unsigned int   u32x4;
typedef __attribute__((ext_vector_type(4)))  int            v4i;
typedef __attribute__((ext_vector_type(4)))  float          f32x4;

// ---------------- CDNA5 async copy (guarded) ----------------
#if defined(__has_builtin)
#if __has_builtin(__builtin_amdgcn_global_load_async_to_lds_b128)
#define HAVE_ASYNC 1
#endif
#endif

#ifdef HAVE_ASYNC
// signature (probe-confirmed via diagnostics): (as1 int4*, as3 int4*, imm offset, imm cpol)
#define ASYNC_CP16(gp, lp, off)                                              \
    __builtin_amdgcn_global_load_async_to_lds_b128(                          \
        (__attribute__((address_space(1))) v4i*)(gp),                        \
        (__attribute__((address_space(3))) v4i*)(lp), (off), 0)
#if __has_builtin(__builtin_amdgcn_s_wait_asynccnt)
#define WAIT_ASYNC() __builtin_amdgcn_s_wait_asynccnt(0)
#else
#define WAIT_ASYNC() asm volatile("s_wait_asynccnt 0" ::: "memory")
#endif
#endif

union Pack16 {
    v16u          v;
    unsigned short s[16];
    u32x4         q[2];
};

__device__ __forceinline__ unsigned short f2bf(float f) {
    unsigned int u = __builtin_bit_cast(unsigned int, f);
    unsigned int r = u + 0x7FFFu + ((u >> 16) & 1u);
    return (unsigned short)(r >> 16);
}

__device__ __forceinline__ v8f wmma_bf16(const Pack16& a, const Pack16& b, v8f c) {
    return __builtin_amdgcn_wmma_f32_16x16x32_bf16(
        false, __builtin_bit_cast(v16bf, a.v),
        false, __builtin_bit_cast(v16bf, b.v),
        (short)0, c, false, false);
}

// block of 256 threads (8 waves): sum-reduce with broadcast
__device__ __forceinline__ float block_sum(float v, volatile float* sm) {
    #pragma unroll
    for (int m = 16; m >= 1; m >>= 1) v += __shfl_xor(v, m, 32);
    int w = threadIdx.x >> 5;
    if ((threadIdx.x & 31) == 0) sm[w] = v;
    __syncthreads();
    float r = sm[0];
    #pragma unroll
    for (int i = 1; i < 8; ++i) r += sm[i];
    __syncthreads();
    return r;
}

// ---------------- cast fp32 -> bf16 ----------------
__global__ void cast_bf16_kernel(const float* __restrict__ src,
                                 unsigned short* __restrict__ dst, int n4) {
    int i = blockIdx.x * blockDim.x + threadIdx.x;
    if (i < n4) {
        f32x4 f = reinterpret_cast<const f32x4*>(src)[i];
        unsigned short o[4] = { f2bf(f.x), f2bf(f.y), f2bf(f.z), f2bf(f.w) };
        reinterpret_cast<unsigned long long*>(dst)[i] =
            *reinterpret_cast<unsigned long long*>(o);
    }
}

// ---------------- LayerNorm + gate ----------------
__global__ __launch_bounds__(256) void ln_gate_kernel(
    const float* __restrict__ hs, const float* __restrict__ ln_g,
    const float* __restrict__ ln_b, const float* __restrict__ gate_w,
    const float* __restrict__ gate_b, float* __restrict__ gates) {
    __shared__ float sm[8];
    int row = blockIdx.x;
    const float* x = hs + (size_t)row * H;
    float s1 = 0.f, s2 = 0.f;
    for (int i = threadIdx.x; i < H; i += 256) {
        float v = x[i]; s1 += v; s2 += v * v;
    }
    s1 = block_sum(s1, sm);
    s2 = block_sum(s2, sm);
    float mu  = s1 / (float)H;
    float var = s2 / (float)H - mu * mu;
    float rstd = rsqrtf(var + 1e-5f);
    float dot = 0.f;
    for (int i = threadIdx.x; i < H; i += 256) {
        float gi = (x[i] - mu) * rstd * ln_g[i] + ln_b[i];
        dot += gi * gate_w[i];
    }
    dot = block_sum(dot, sm);
    if (threadIdx.x == 0) {
        float z = dot + gate_b[0];
        z = fminf(10.f, fmaxf(-10.f, z));
        gates[row] = 1.f / (1.f + __expf(-z));
    }
}

// ---------------- regularization loss (scalar) ----------------
__global__ __launch_bounds__(256) void reg_kernel(const float* __restrict__ gates,
                                                  float* __restrict__ out) {
    __shared__ float sm[8];
    float a = 0.f, b = 0.f, c = 0.f;
    for (int i = threadIdx.x; i < S; i += 256) {
        float g  = gates[i];
        float gc = fminf(1.f - 1e-5f, fmaxf(1e-5f, g));
        a += g * (1.f - g);
        b += g * __logf(gc) + (1.f - g) * __logf(1.f - gc);
        c += g;
    }
    a = block_sum(a, sm);
    b = block_sum(b, sm);
    c = block_sum(c, sm);
    if (threadIdx.x == 0)
        out[0] = (-0.1f * a - 0.01f * b + 0.1f * c) / (float)S;
}

// ---------------- tiled WMMA GEMM: C = A(MxK) * B(KxN), bf16 in ----------------
// block tile 128(M) x 64(N), k-step 32; 8 waves, one 16x64 strip each.
__global__ __launch_bounds__(256) void gemm_bf16_kernel(
    const unsigned short* __restrict__ A, const unsigned short* __restrict__ B,
    unsigned short* __restrict__ Cb, float* __restrict__ Cf,
    int M, int N, int K) {
    __shared__ __align__(16) unsigned short As[128][32];   // 8 KB
    __shared__ __align__(16) unsigned short Bts[64][32];   // transposed B, 4 KB

    int bm = blockIdx.x * 128;
    int bn = blockIdx.y * 64;
    int tid = threadIdx.x;
    int wave = tid >> 5;
    int lane = tid & 31;
    int half = lane >> 4;
    int l16  = lane & 15;

    v8f acc[4] = {};

    for (int k0 = 0; k0 < K; k0 += 32) {
        // stage A tile 128x32: async memory->LDS (no VGPR staging) if available
        #pragma unroll
        for (int it = 0; it < 2; ++it) {
            int c  = tid + it * 256;
            int r  = c >> 2;
            int cc = (c & 3) * 8;
#ifdef HAVE_ASYNC
            ASYNC_CP16(&A[(size_t)(bm + r) * K + k0 + cc], &As[r][cc], 0);
#else
            *reinterpret_cast<u32x4*>(&As[r][cc]) =
                *reinterpret_cast<const u32x4*>(&A[(size_t)(bm + r) * K + k0 + cc]);
#endif
        }
        // stage B tile 32x64 transposed -> Bts[col][k] (manual: needs transpose)
        {
            int c  = tid;            // 256 chunks of 8
            int kk = c >> 3;
            int cc = (c & 7) * 8;
            union { u32x4 q; unsigned short s[8]; } t;
            t.q = *reinterpret_cast<const u32x4*>(&B[(size_t)(k0 + kk) * N + bn + cc]);
            #pragma unroll
            for (int i = 0; i < 8; ++i) Bts[cc + i][kk] = t.s[i];
        }
        if (k0 + 32 < K)
            __builtin_prefetch(&A[(size_t)(bm + (tid >> 1)) * K + k0 + 32], 0, 0);
#ifdef HAVE_ASYNC
        WAIT_ASYNC();
#endif
        __syncthreads();

        // A operand: rows wave*16..+15 per documented 16-bit A layout
        Pack16 pa;
        {
            int row = wave * 16 + l16;
            int kb  = half * 8;
            pa.q[0] = *reinterpret_cast<const u32x4*>(&As[row][kb]);
            pa.q[1] = *reinterpret_cast<const u32x4*>(&As[row][kb + 16]);
        }
        #pragma unroll
        for (int nt = 0; nt < 4; ++nt) {
            Pack16 pb;
            int col = nt * 16 + l16;
            int kb  = half * 8;
            pb.q[0] = *reinterpret_cast<const u32x4*>(&Bts[col][kb]);
            pb.q[1] = *reinterpret_cast<const u32x4*>(&Bts[col][kb + 16]);
            acc[nt] = wmma_bf16(pa, pb, acc[nt]);
        }
        __syncthreads();
    }

    #pragma unroll
    for (int nt = 0; nt < 4; ++nt) {
        #pragma unroll
        for (int r = 0; r < 8; ++r) {
            int row = bm + wave * 16 + r + 8 * half;
            int col = bn + nt * 16 + l16;
            float v = acc[nt][r];
            if (Cb) Cb[(size_t)row * N + col] = f2bf(v);
            else    Cf[(size_t)row * N + col] = v;
        }
    }
}

// ---------------- fused dual-softmax flash attention ----------------
// grid: (S/16 q-blocks, NH heads), 1 wave per block.
__global__ __launch_bounds__(32) void attn_kernel(
    const unsigned short* __restrict__ Q,   // [S][H] bf16
    const unsigned short* __restrict__ Km,  // [S][KVW] bf16
    const unsigned short* __restrict__ Vm,  // [S][KVW] bf16
    const float* __restrict__ gates,        // [S]
    unsigned short* __restrict__ AO) {      // [S][H] bf16
    __shared__ __align__(16) unsigned short Vs[32][HD];    // 8 KB
    __shared__ __align__(16) unsigned short Ps[2][16][32]; // probs (local, global)

    int qb  = blockIdx.x;
    int h   = blockIdx.y;
    int kvh = h & (KVH - 1);
    int q0  = qb * 16;
    int lane = threadIdx.x;
    int half = lane >> 4, l16 = lane & 15;

    // Q tile -> A operands for 4 k-chunks of 32 over HD
    Pack16 qa[4];
    {
        const unsigned short* qrow = &Q[(size_t)(q0 + l16) * H + h * HD];
        #pragma unroll
        for (int c = 0; c < 4; ++c) {
            int kb = c * 32 + half * 8;
            qa[c].q[0] = *reinterpret_cast<const u32x4*>(qrow + kb);
            qa[c].q[1] = *reinterpret_cast<const u32x4*>(qrow + kb + 16);
        }
    }

    v8f accL[8] = {}, accG[8] = {};
    float mL[8], mG[8], lL[8], lG[8];
    #pragma unroll
    for (int r = 0; r < 8; ++r) { mL[r] = -3e38f; mG[r] = -3e38f; lL[r] = 0.f; lG[r] = 0.f; }

    const float scl = 0.08838834764831845f;  // 1/sqrt(HD)
    int jend = (S < q0 + 16 + WINH) ? S : (q0 + 16 + WINH);

    for (int j0 = 0; j0 < jend; j0 += 32) {
        __syncthreads();
        // stage V tile (row-major): one key row per lane, 256B each
        {
            const unsigned short* vr = &Vm[(size_t)(j0 + lane) * KVW + kvh * HD];
#ifdef HAVE_ASYNC
            void* lp = (void*)&Vs[lane][0];
            ASYNC_CP16(vr, lp, 0);    ASYNC_CP16(vr, lp, 16);
            ASYNC_CP16(vr, lp, 32);   ASYNC_CP16(vr, lp, 48);
            ASYNC_CP16(vr, lp, 64);   ASYNC_CP16(vr, lp, 80);
            ASYNC_CP16(vr, lp, 96);   ASYNC_CP16(vr, lp, 112);
            ASYNC_CP16(vr, lp, 128);  ASYNC_CP16(vr, lp, 144);
            ASYNC_CP16(vr, lp, 160);  ASYNC_CP16(vr, lp, 176);
            ASYNC_CP16(vr, lp, 192);  ASYNC_CP16(vr, lp, 208);
            ASYNC_CP16(vr, lp, 224);  ASYNC_CP16(vr, lp, 240);
#else
            #pragma unroll
            for (int i = 0; i < 16; ++i)
                *reinterpret_cast<u32x4*>(&Vs[lane][i * 8]) =
                    *reinterpret_cast<const u32x4*>(vr + i * 8);
#endif
        }

        // scores: 2 n-tiles of 16 keys, K operand straight from global (contiguous)
        v8f s[2] = {};
        #pragma unroll
        for (int t = 0; t < 2; ++t) {
            const unsigned short* kr = &Km[(size_t)(j0 + t * 16 + l16) * KVW + kvh * HD];
            #pragma unroll
            for (int c = 0; c < 4; ++c) {
                Pack16 pb;
                pb.q[0] = *reinterpret_cast<const u32x4*>(kr + c * 32 + half * 8);
                pb.q[1] = *reinterpret_cast<const u32x4*>(kr + c * 32 + half * 8 + 16);
                s[t] = wmma_bf16(qa[c], pb, s[t]);
            }
        }

        // masking + dual online softmax
        float svL[2][8], svG[2][8], rmL[8], rmG[8];
        #pragma unroll
        for (int r = 0; r < 8; ++r) { rmL[r] = -3e38f; rmG[r] = -3e38f; }
        #pragma unroll
        for (int t = 0; t < 2; ++t) {
            #pragma unroll
            for (int r = 0; r < 8; ++r) {
                int i = q0 + r + 8 * half;
                int j = j0 + t * 16 + l16;
                float sv = s[t][r] * scl;
                int d = i - j;
                float vG = (j <= i) ? sv : -3e38f;
                float vL = (d <= WINH && d >= -WINH) ? sv : -3e38f;
                svG[t][r] = vG; svL[t][r] = vL;
                rmG[r] = fmaxf(rmG[r], vG);
                rmL[r] = fmaxf(rmL[r], vL);
            }
        }
        #pragma unroll
        for (int r = 0; r < 8; ++r) {
            #pragma unroll
            for (int m = 1; m <= 8; m <<= 1) {
                rmG[r] = fmaxf(rmG[r], __shfl_xor(rmG[r], m, 32));
                rmL[r] = fmaxf(rmL[r], __shfl_xor(rmL[r], m, 32));
            }
        }
        float psG[8], psL[8];
        #pragma unroll
        for (int r = 0; r < 8; ++r) {
            float mGn = fmaxf(mG[r], rmG[r]);
            float mLn = fmaxf(mL[r], rmL[r]);
            float cG = __expf(mG[r] - mGn);
            float cL = __expf(mL[r] - mLn);
            mG[r] = mGn; mL[r] = mLn;
            lG[r] *= cG; lL[r] *= cL;
            #pragma unroll
            for (int nt = 0; nt < 8; ++nt) { accG[nt][r] *= cG; accL[nt][r] *= cL; }
            float sg = 0.f, sl = 0.f;
            #pragma unroll
            for (int t = 0; t < 2; ++t) {
                float pg = (svG[t][r] > -1e38f) ? __expf(svG[t][r] - mGn) : 0.f;
                float pl = (svL[t][r] > -1e38f) ? __expf(svL[t][r] - mLn) : 0.f;
                Ps[0][r + 8 * half][t * 16 + l16] = f2bf(pl);
                Ps[1][r + 8 * half][t * 16 + l16] = f2bf(pg);
                sg += pg; sl += pl;
            }
            psG[r] = sg; psL[r] = sl;
        }
        #pragma unroll
        for (int r = 0; r < 8; ++r) {
            #pragma unroll
            for (int m = 1; m <= 8; m <<= 1) {
                psG[r] += __shfl_xor(psG[r], m, 32);
                psL[r] += __shfl_xor(psL[r], m, 32);
            }
            lG[r] += psG[r]; lL[r] += psL[r];
        }
#ifdef HAVE_ASYNC
        WAIT_ASYNC();
#endif
        __syncthreads();

        // probs -> A operands (transposed through LDS)
        Pack16 paL, paG;
        paL.q[0] = *reinterpret_cast<const u32x4*>(&Ps[0][l16][half * 8]);
        paL.q[1] = *reinterpret_cast<const u32x4*>(&Ps[0][l16][half * 8 + 16]);
        paG.q[0] = *reinterpret_cast<const u32x4*>(&Ps[1][l16][half * 8]);
        paG.q[1] = *reinterpret_cast<const u32x4*>(&Ps[1][l16][half * 8 + 16]);

        // P @ V : shared V B-operand feeds both paths
        #pragma unroll
        for (int nt = 0; nt < 8; ++nt) {
            Pack16 pv;
            #pragma unroll
            for (int u = 0; u < 16; ++u)
                pv.s[u] = Vs[half * 8 + u + (u & 8)][nt * 16 + l16];
            accL[nt] = wmma_bf16(paL, pv, accL[nt]);
            accG[nt] = wmma_bf16(paG, pv, accG[nt]);
        }
    }

    // gate-blend, normalize, store attn_out (bf16)
    #pragma unroll
    for (int r = 0; r < 8; ++r) {
        int row = q0 + r + 8 * half;
        float g = gates[row];
        float iL = 1.f / lL[r];
        float iG = 1.f / lG[r];
        #pragma unroll
        for (int nt = 0; nt < 8; ++nt) {
            float o = g * accL[nt][r] * iL + (1.f - g) * accG[nt][r] * iG;
            AO[(size_t)row * H + h * HD + nt * 16 + l16] = f2bf(o);
        }
    }
}

// ---------------- launch ----------------
extern "C" void kernel_launch(void* const* d_in, const int* in_sizes, int n_in,
                              void* d_out, int out_size, void* d_ws, size_t ws_size,
                              hipStream_t stream) {
    const float* hs     = (const float*)d_in[0];
    const float* Wq     = (const float*)d_in[1];
    const float* Wk     = (const float*)d_in[2];
    const float* Wv     = (const float*)d_in[3];
    const float* Wo     = (const float*)d_in[4];
    const float* gate_w = (const float*)d_in[5];
    const float* gate_b = (const float*)d_in[6];
    const float* ln_g   = (const float*)d_in[7];
    const float* ln_b   = (const float*)d_in[8];
    float* out = (float*)d_out;

    char* w = (char*)d_ws;
    auto take = [&](size_t bytes) { char* p = w; w += (bytes + 255) & ~size_t(255); return p; };
    unsigned short* hsb = (unsigned short*)take((size_t)S * H * 2);
    unsigned short* Wqb = (unsigned short*)take((size_t)H * H * 2);
    unsigned short* Wkb = (unsigned short*)take((size_t)H * KVW * 2);
    unsigned short* Wvb = (unsigned short*)take((size_t)H * KVW * 2);
    unsigned short* Wob = (unsigned short*)take((size_t)H * H * 2);
    unsigned short* Qb  = (unsigned short*)take((size_t)S * H * 2);
    unsigned short* Kb  = (unsigned short*)take((size_t)S * KVW * 2);
    unsigned short* Vb  = (unsigned short*)take((size_t)S * KVW * 2);
    unsigned short* AOb = (unsigned short*)take((size_t)S * H * 2);
    float* gates        = (float*)take((size_t)S * 4);

    // casts to bf16
    cast_bf16_kernel<<<(S * H / 4) / 256, 256, 0, stream>>>(hs, hsb, S * H / 4);
    cast_bf16_kernel<<<(H * H / 4) / 256, 256, 0, stream>>>(Wq, Wqb, H * H / 4);
    cast_bf16_kernel<<<(H * KVW / 4) / 256, 256, 0, stream>>>(Wk, Wkb, H * KVW / 4);
    cast_bf16_kernel<<<(H * KVW / 4) / 256, 256, 0, stream>>>(Wv, Wvb, H * KVW / 4);
    cast_bf16_kernel<<<(H * H / 4) / 256, 256, 0, stream>>>(Wo, Wob, H * H / 4);

    // LN + gates, reg loss scalar
    ln_gate_kernel<<<S, 256, 0, stream>>>(hs, ln_g, ln_b, gate_w, gate_b, gates);
    reg_kernel<<<1, 256, 0, stream>>>(gates, out + (size_t)S * H);

    // projections (bf16 out)
    gemm_bf16_kernel<<<dim3(S / 128, H / 64), 256, 0, stream>>>(
        hsb, Wqb, Qb, nullptr, S, H, H);
    gemm_bf16_kernel<<<dim3(S / 128, KVW / 64), 256, 0, stream>>>(
        hsb, Wkb, Kb, nullptr, S, KVW, H);
    gemm_bf16_kernel<<<dim3(S / 128, KVW / 64), 256, 0, stream>>>(
        hsb, Wvb, Vb, nullptr, S, KVW, H);

    // fused gated dual-softmax attention
    attn_kernel<<<dim3(S / 16, NH), 32, 0, stream>>>(Qb, Kb, Vb, gates, AOb);

    // output projection (f32 out -> d_out)
    gemm_bf16_kernel<<<dim3(S / 128, H / 64), 256, 0, stream>>>(
        AOb, Wob, nullptr, out, S, H, H);
}
